// MulAdapter_61735859912854
// MI455X (gfx1250) — matrix operand order repaired and checked by reading, not compiled, not run on hardware
//
#include <hip/hip_runtime.h>
#include <hip/hip_bf16.h>

typedef __bf16 bf16;
typedef __attribute__((ext_vector_type(16))) __bf16 v16bf;
typedef __attribute__((ext_vector_type(8)))  __bf16 v8bf;
typedef __attribute__((ext_vector_type(8)))  float  v8f;

#define BATCH 2
#define CH    256
#define NTOK  4096

__device__ __forceinline__ v8f wmma_bf16(v16bf a, v16bf b, v8f c) {
    return __builtin_amdgcn_wmma_f32_16x16x32_bf16(false, a, false, b, (short)0, c, false, false);
}

__device__ __forceinline__ v8f zero8() {
    v8f z = {0.f,0.f,0.f,0.f,0.f,0.f,0.f,0.f};
    return z;
}

// A operand fragment (16x32 bf16, M rows / K cols). `p0` points at element (m0,k0);
// rows are `stride` elements apart, K contiguous.
// ISA layout: lane<16 -> halves[0..7]=K0..7, halves[8..15]=K16..23; lane>=16 shifted by 8.
__device__ __forceinline__ v16bf frag_a(const bf16* p0, int stride) {
    int lane = threadIdx.x & 31;
    const bf16* p = p0 + (lane & 15) * stride + ((lane >> 4) << 3);
    v8bf lo = *(const v8bf*)(p);
    v8bf hi = *(const v8bf*)(p + 16);
    return __builtin_shufflevector(lo, hi, 0,1,2,3,4,5,6,7,8,9,10,11,12,13,14,15);
}

// B operand fragment (32x16 bf16, K rows / N cols) sourced from an [N][K] (K-contiguous)
// layout. lane<16 -> halves[0..7]=K0..7, halves[8..15]=K16..23; lane>=16 shifted by 8.
__device__ __forceinline__ v16bf frag_b(const bf16* p0, int stride) {
    int lane = threadIdx.x & 31;
    const bf16* p = p0 + (lane & 15) * stride + ((lane >> 4) << 3);
    v8bf lo = *(const v8bf*)(p);
    v8bf hi = *(const v8bf*)(p + 16);
    return __builtin_shufflevector(lo, hi, 0,1,2,3,4,5,6,7,8,9,10,11,12,13,14,15);
}

// ---------------- prep: f32 -> bf16 conversions ----------------
__global__ __launch_bounds__(256) void k_cvt(const float* in, bf16* out, int n) {
    int i = blockIdx.x * 256 + threadIdx.x;
    if (i < n) out[i] = (bf16)in[i];
}

// x [B,C,N] f32 -> x_nc [B,N,C] bf16 (transpose + convert)
__global__ __launch_bounds__(256) void k_x_to_nc(const float* x, bf16* x_nc) {
    int i = blockIdx.x * 256 + threadIdx.x;             // over B*C*N, n fastest
    int n = i & (NTOK - 1);
    int c = (i >> 12) & (CH - 1);
    int b = i >> 20;
    x_nc[(((size_t)b * NTOK + n) << 8) + c] = (bf16)x[i];
}

// ---------------- kernel 1: fused conv1(+bias+relu) -> conv2(+bias) ----------------
// y1 intermediate lives entirely in LDS. Outputs y in [N,C] layout (f32 for the
// residual, bf16 for the QKV GEMM B-operand).
__global__ __launch_bounds__(256) void k_conv(const bf16* __restrict__ x_nc,
                                              const bf16* __restrict__ w1, const float* __restrict__ b1,
                                              const bf16* __restrict__ w2, const float* __restrict__ b2,
                                              float* __restrict__ y_f, bf16* __restrict__ y_bf) {
    __shared__ bf16 y1[32][264];                        // [n][c], padded, 16B-aligned rows
    const int b = blockIdx.y, n0 = blockIdx.x * 32;
    const int wave = threadIdx.x >> 5, lane = threadIdx.x & 31;
    const int col = lane & 15, hi = lane >> 4;
    const bf16* X = x_nc + (((size_t)b * NTOK + n0) << 8);

    // GEMM1: y1 = relu(w1 * x + b1); wave owns m-tiles {2w,2w+1} x n-tiles {0,1}
    v8f acc[2][2];
    for (int mt = 0; mt < 2; ++mt) for (int nt = 0; nt < 2; ++nt) acc[mt][nt] = zero8();
    for (int kt = 0; kt < 8; ++kt) {
        v16bf bf0 = frag_b(X + 0 * 16 * CH + kt * 32, CH);
        v16bf bf1 = frag_b(X + 1 * 16 * CH + kt * 32, CH);
        for (int mt = 0; mt < 2; ++mt) {
            v16bf a = frag_a(w1 + (2 * wave + mt) * 16 * CH + kt * 32, CH);
            acc[mt][0] = wmma_bf16(a, bf0, acc[mt][0]);
            acc[mt][1] = wmma_bf16(a, bf1, acc[mt][1]);
        }
    }
    for (int mt = 0; mt < 2; ++mt) {
        int m0 = (2 * wave + mt) * 16;
        v8f bb = *(const v8f*)&b1[m0 + 8 * hi];
        for (int nt = 0; nt < 2; ++nt) {
            v8bf pk;
#pragma unroll
            for (int r = 0; r < 8; ++r) pk[r] = (bf16)fmaxf(acc[mt][nt][r] + bb[r], 0.0f);
            *(v8bf*)&y1[nt * 16 + col][m0 + 8 * hi] = pk;
        }
    }
    __syncthreads();

    // GEMM2: y = w2 * y1 + b2, B fragments straight out of LDS
    for (int mt = 0; mt < 2; ++mt) for (int nt = 0; nt < 2; ++nt) acc[mt][nt] = zero8();
    for (int kt = 0; kt < 8; ++kt) {
        v16bf bf0 = frag_b(&y1[0][0]  + kt * 32, 264);
        v16bf bf1 = frag_b(&y1[16][0] + kt * 32, 264);
        for (int mt = 0; mt < 2; ++mt) {
            v16bf a = frag_a(w2 + (2 * wave + mt) * 16 * CH + kt * 32, CH);
            acc[mt][0] = wmma_bf16(a, bf0, acc[mt][0]);
            acc[mt][1] = wmma_bf16(a, bf1, acc[mt][1]);
        }
    }
    for (int mt = 0; mt < 2; ++mt) {
        int c0 = (2 * wave + mt) * 16;
        v8f bb = *(const v8f*)&b2[c0 + 8 * hi];
        for (int nt = 0; nt < 2; ++nt) {
            int n = n0 + nt * 16 + col;
            v8f vf; v8bf pk;
#pragma unroll
            for (int r = 0; r < 8; ++r) { float f = acc[mt][nt][r] + bb[r]; vf[r] = f; pk[r] = (bf16)f; }
            size_t base = (((size_t)b * NTOK + n) << 8) + c0 + 8 * hi;
            *(v8f*)&y_f[base]  = vf;
            *(v8bf*)&y_bf[base] = pk;
        }
    }
}

// ---------------- kernel 2: fused QKV (q^T = wq*y etc.) ----------------
// q,k written [N,C]; v written [C,N] (A-operand layout for the output GEMM).
__global__ __launch_bounds__(256) void k_qkv(const bf16* __restrict__ y_bf,
                                             const bf16* __restrict__ wq, const float* __restrict__ bq,
                                             const bf16* __restrict__ wk, const float* __restrict__ bk,
                                             const bf16* __restrict__ wv, const float* __restrict__ bv,
                                             bf16* __restrict__ q_nc, bf16* __restrict__ k_nc,
                                             bf16* __restrict__ v_cn) {
    const int b = blockIdx.y, n0 = blockIdx.x * 32;
    const int wave = threadIdx.x >> 5, lane = threadIdx.x & 31;
    const int col = lane & 15, hi = lane >> 4;
    const bf16* Y = y_bf + (((size_t)b * NTOK + n0) << 8);
    const bf16* W[3] = {wq, wk, wv};
    const float* BI[3] = {bq, bk, bv};

    v8f acc[3][2][2];
    for (int s = 0; s < 3; ++s) for (int mt = 0; mt < 2; ++mt) for (int nt = 0; nt < 2; ++nt)
        acc[s][mt][nt] = zero8();
    for (int kt = 0; kt < 8; ++kt) {
        v16bf bf0 = frag_b(Y + 0 * 16 * CH + kt * 32, CH);
        v16bf bf1 = frag_b(Y + 1 * 16 * CH + kt * 32, CH);
        for (int s = 0; s < 3; ++s)
            for (int mt = 0; mt < 2; ++mt) {
                v16bf a = frag_a(W[s] + (2 * wave + mt) * 16 * CH + kt * 32, CH);
                acc[s][mt][0] = wmma_bf16(a, bf0, acc[s][mt][0]);
                acc[s][mt][1] = wmma_bf16(a, bf1, acc[s][mt][1]);
            }
    }
    for (int s = 0; s < 3; ++s)
        for (int mt = 0; mt < 2; ++mt) {
            int o0 = (2 * wave + mt) * 16;
            v8f bb = *(const v8f*)&BI[s][o0 + 8 * hi];
            for (int nt = 0; nt < 2; ++nt) {
                int n = n0 + nt * 16 + col;
                v8bf pk;
#pragma unroll
                for (int r = 0; r < 8; ++r) pk[r] = (bf16)(acc[s][mt][nt][r] + bb[r]);
                if (s == 0)      *(v8bf*)&q_nc[(((size_t)b * NTOK + n) << 8) + o0 + 8 * hi] = pk;
                else if (s == 1) *(v8bf*)&k_nc[(((size_t)b * NTOK + n) << 8) + o0 + 8 * hi] = pk;
                else {
#pragma unroll
                    for (int r = 0; r < 8; ++r)
                        v_cn[(((size_t)b * CH + o0 + 8 * hi + r) << 12) + n] = pk[r];
                }
            }
        }
}

// ---------------- kernel 3: softmax row stats (online max/sum over m) ----------------
__global__ __launch_bounds__(256) void k_stats(const bf16* __restrict__ q_nc,
                                               const bf16* __restrict__ k_nc,
                                               float* __restrict__ rowmax, float* __restrict__ rowinv) {
    const int b = blockIdx.y, n0 = blockIdx.x * 16;
    const int wave = threadIdx.x >> 5, lane = threadIdx.x & 31, hi = lane >> 4;
    const bf16* Q = q_nc + (((size_t)b * NTOK + n0) << 8);
    const bf16* K = k_nc + ((size_t)b * NTOK << 8);

    v16bf afr[8];
#pragma unroll
    for (int kt = 0; kt < 8; ++kt) afr[kt] = frag_a(Q + kt * 32, CH);

    float mi[8], si[8];
#pragma unroll
    for (int r = 0; r < 8; ++r) { mi[r] = -1.0e30f; si[r] = 0.0f; }

    for (int mt = wave; mt < NTOK / 16; mt += 8) {
        v8f d = zero8();
#pragma unroll
        for (int kt = 0; kt < 8; ++kt)
            d = wmma_bf16(afr[kt], frag_b(K + mt * 16 * CH + kt * 32, CH), d);
#pragma unroll
        for (int r = 0; r < 8; ++r) {
            float v = d[r] * 16.0f;                       // * sqrt(C)
            float mn = fmaxf(mi[r], v);
            si[r] = si[r] * __expf(mi[r] - mn) + __expf(v - mn);
            mi[r] = mn;
        }
    }
    // reduce across the 16 lanes sharing the same score rows
    for (int off = 1; off < 16; off <<= 1) {
#pragma unroll
        for (int r = 0; r < 8; ++r) {
            float mo = __shfl_xor(mi[r], off, 32);
            float so = __shfl_xor(si[r], off, 32);
            float mn = fmaxf(mi[r], mo);
            si[r] = si[r] * __expf(mi[r] - mn) + so * __expf(mo - mn);
            mi[r] = mn;
        }
    }
    __shared__ float smax[16][8], ssum[16][8];
    if ((lane & 15) == 0) {
#pragma unroll
        for (int r = 0; r < 8; ++r) { smax[8 * hi + r][wave] = mi[r]; ssum[8 * hi + r][wave] = si[r]; }
    }
    __syncthreads();
    if (threadIdx.x < 16) {
        float m = smax[threadIdx.x][0], s = ssum[threadIdx.x][0];
        for (int w = 1; w < 8; ++w) {
            float mo = smax[threadIdx.x][w], so = ssum[threadIdx.x][w];
            float mn = fmaxf(m, mo);
            s = s * __expf(m - mn) + so * __expf(mo - mn);
            m = mn;
        }
        rowmax[b * NTOK + n0 + threadIdx.x] = m;
        rowinv[b * NTOK + n0 + threadIdx.x] = 1.0f / s;
    }
}

// ---------------- kernel 4: recompute scores -> p -> out = v^T * p + y ----------------
__global__ __launch_bounds__(256) void k_out(const bf16* __restrict__ q_nc, const bf16* __restrict__ k_nc,
                                             const bf16* __restrict__ v_cn,
                                             const float* __restrict__ rowmax, const float* __restrict__ rowinv,
                                             const float* __restrict__ y_f, float* __restrict__ out) {
    __shared__ bf16 pB[64][40];                          // p tile, [m][n] B-fragment layout
    const int b = blockIdx.y, m0 = blockIdx.x * 64;
    const int wave = threadIdx.x >> 5, lane = threadIdx.x & 31;
    const int col = lane & 15, hi = lane >> 4;
    const int mtA = wave & 3, nsA = wave >> 2;           // phase-A role: score tile (nsA, mtA)

    const bf16* Qb = q_nc + ((size_t)b * NTOK << 8);
    const bf16* K  = k_nc + (((size_t)b * NTOK + m0 + mtA * 16) << 8);
    const bf16* V  = v_cn + ((size_t)b * CH << 12);
    const float* RM = rowmax + b * NTOK;
    const float* RI = rowinv + b * NTOK;

    v16bf bfrA[8];                                        // k-tile for this wave's m columns: loop-invariant
#pragma unroll
    for (int kt = 0; kt < 8; ++kt) bfrA[kt] = frag_b(K + kt * 32, CH);

    v8f acc[2][4];
    for (int ci = 0; ci < 2; ++ci) for (int mt = 0; mt < 4; ++mt) acc[ci][mt] = zero8();

    for (int ns = 0; ns < NTOK / 32; ++ns) {
        int nbase = ns * 32;
        // phase A: one 16x16 score tile per wave -> p -> LDS
        {
            const bf16* Qt = Qb + ((size_t)(nbase + nsA * 16) << 8);
            v8f d = zero8();
#pragma unroll
            for (int kt = 0; kt < 8; ++kt) d = wmma_bf16(frag_a(Qt + kt * 32, CH), bfrA[kt], d);
            v8bf pk;
#pragma unroll
            for (int r = 0; r < 8; ++r) {
                int n = nbase + nsA * 16 + 8 * hi + r;
                pk[r] = (bf16)(__expf(d[r] * 16.0f - RM[n]) * RI[n]);
            }
            *(v8bf*)&pB[mtA * 16 + col][nsA * 16 + 8 * hi] = pk;
        }
        __syncthreads();
        // phase B: out[c, m] += v^T[c, nchunk] * p[nchunk, m]
        for (int ci = 0; ci < 2; ++ci) {
            int c0 = (wave * 2 + ci) * 16;
            const bf16* vp = V + ((size_t)c0 << 12) + nbase;
            __builtin_prefetch((const void*)(vp + 32), 0, 1);   // global_prefetch_b8 for next K chunk
            v16bf a = frag_a(vp, NTOK);
#pragma unroll
            for (int mt = 0; mt < 4; ++mt)
                acc[ci][mt] = wmma_bf16(a, frag_b(&pB[mt * 16][0], 40), acc[ci][mt]);
        }
        __syncthreads();
    }
    // epilogue: residual add (y in [N,C] -> contiguous v8f per lane), store [B,C,N]
    for (int ci = 0; ci < 2; ++ci) {
        int c0 = (wave * 2 + ci) * 16;
        for (int mt = 0; mt < 4; ++mt) {
            int m = m0 + mt * 16 + col;
            v8f y8 = *(const v8f*)&y_f[(((size_t)b * NTOK + m) << 8) + c0 + 8 * hi];
#pragma unroll
            for (int r = 0; r < 8; ++r)
                out[(((size_t)b * CH + c0 + 8 * hi + r) << 12) + m] = acc[ci][mt][r] + y8[r];
        }
    }
}

// ---------------- host ----------------
extern "C" void kernel_launch(void* const* d_in, const int* in_sizes, int n_in,
                              void* d_out, int out_size, void* d_ws, size_t ws_size,
                              hipStream_t stream) {
    (void)in_sizes; (void)n_in; (void)out_size; (void)ws_size;
    const float* x  = (const float*)d_in[0];
    const float* w1 = (const float*)d_in[1];  const float* b1 = (const float*)d_in[2];
    const float* w2 = (const float*)d_in[3];  const float* b2 = (const float*)d_in[4];
    const float* wq = (const float*)d_in[5];  const float* bq = (const float*)d_in[6];
    const float* wk = (const float*)d_in[7];  const float* bk = (const float*)d_in[8];
    const float* wv = (const float*)d_in[9];  const float* bv = (const float*)d_in[10];
    float* out = (float*)d_out;

    char* ws = (char*)d_ws;
    size_t off = 0;
    auto carve = [&](size_t bytes) { char* p = ws + off; off += (bytes + 255) & ~(size_t)255; return p; };
    const size_t BNC = (size_t)BATCH * NTOK * CH;
    bf16*  x_nc = (bf16*)carve(BNC * 2);
    bf16*  w1b  = (bf16*)carve(CH * CH * 2);
    bf16*  w2b  = (bf16*)carve(CH * CH * 2);
    bf16*  wqb  = (bf16*)carve(CH * CH * 2);
    bf16*  wkb  = (bf16*)carve(CH * CH * 2);
    bf16*  wvb  = (bf16*)carve(CH * CH * 2);
    bf16*  y_bf = (bf16*)carve(BNC * 2);
    float* y_f  = (float*)carve(BNC * 4);
    bf16*  q_nc = (bf16*)carve(BNC * 2);
    bf16*  k_nc = (bf16*)carve(BNC * 2);
    bf16*  v_cn = (bf16*)carve(BNC * 2);
    float* rmax = (float*)carve((size_t)BATCH * NTOK * 4);
    float* rinv = (float*)carve((size_t)BATCH * NTOK * 4);

    k_x_to_nc<<<(int)(BNC / 256), 256, 0, stream>>>(x, x_nc);
    k_cvt<<<CH * CH / 256, 256, 0, stream>>>(w1, w1b, CH * CH);
    k_cvt<<<CH * CH / 256, 256, 0, stream>>>(w2, w2b, CH * CH);
    k_cvt<<<CH * CH / 256, 256, 0, stream>>>(wq, wqb, CH * CH);
    k_cvt<<<CH * CH / 256, 256, 0, stream>>>(wk, wkb, CH * CH);
    k_cvt<<<CH * CH / 256, 256, 0, stream>>>(wv, wvb, CH * CH);

    k_conv<<<dim3(NTOK / 32, BATCH), 256, 0, stream>>>(x_nc, w1b, b1, w2b, b2, y_f, y_bf);
    k_qkv<<<dim3(NTOK / 32, BATCH), 256, 0, stream>>>(y_bf, wqb, bq, wkb, bk, wvb, bv, q_nc, k_nc, v_cn);
    k_stats<<<dim3(NTOK / 16, BATCH), 256, 0, stream>>>(q_nc, k_nc, rmax, rinv);
    k_out<<<dim3(NTOK / 64, BATCH), 256, 0, stream>>>(q_nc, k_nc, v_cn, rmax, rinv, y_f, out);
}
